// HybridQLSTM_65481071407212
// MI455X (gfx1250) — matrix operand order repaired and checked by reading, not compiled, not run on hardware
//
#include <hip/hip_runtime.h>
#include <hip/hip_bf16.h>
#include <math.h>

typedef __attribute__((ext_vector_type(16))) __bf16 v16bf;
typedef __attribute__((ext_vector_type(8)))  __bf16 v8bf;
typedef __attribute__((ext_vector_type(8)))  float  v8f;
typedef int v4i __attribute__((vector_size(16)));   // matches async-LDS builtin param

constexpr int T = 128, B = 256, D = 1024, H = 1024, NQ = 8, TAGS = 1024;
constexpr int C  = D + H;        // 2048 (comb width)
constexpr int GN = 4 * NQ;       // 32  (gate-gemm N, also padded K of proj gemm)

// ---------------- WMMA helpers ----------------

__device__ inline v8f wmma_bf16(v16bf a, v16bf b, v8f c) {
  // D = A(16x32 bf16) * B(32x16 bf16) + C(16x16 f32)
  return __builtin_amdgcn_wmma_f32_16x16x32_bf16(false, a, false, b, (short)0, c,
                                                 false, false);
}

// Load one 16x32 bf16 A tile from a row-major matrix (lda elements per row).
// Layout: lane m (0..15): K0..15 ; lane m+16: K16..31 (matches packed B).
__device__ inline v16bf load_a_rowmajor(const __bf16* tile00, int lda, int lane) {
  int m  = lane & 15;
  int kb = (lane & 16) ? 16 : 0;
  const __bf16* r = tile00 + m * lda + kb;
  union { v16bf v; v8bf h[2]; } u;
  u.h[0] = *(const v8bf*)(r);
  u.h[1] = *(const v8bf*)(r + 8);
  return u.v;
}

// B tiles are pre-packed per-lane: tile = 32 lanes x 16 halves (1024 bytes).
__device__ inline v16bf load_b_packed(const __bf16* tile, int lane) {
  return *(const v16bf*)(tile + lane * 16);
}

__device__ inline float sigmoidf_(float x) { return 1.f / (1.f + __expf(-x)); }

// -------- gfx1250 async global->LDS staging (ASYNCcnt-tracked) --------

#if __has_builtin(__builtin_amdgcn_global_load_async_to_lds_b128)
#define HAVE_ASYNC_LDS 1
#else
#define HAVE_ASYNC_LDS 0
#endif
#if __has_builtin(__builtin_amdgcn_s_wait_asynccnt)
#define HAVE_WAIT_ASYNC 1
#else
#define HAVE_WAIT_ASYNC 0
#endif

__device__ inline void async_copy16(const void* gsrc, void* lds_dst) {
#if HAVE_ASYNC_LDS
  __builtin_amdgcn_global_load_async_to_lds_b128(
      (__attribute__((address_space(1))) v4i*)(void*)gsrc,
      (__attribute__((address_space(3))) v4i*)lds_dst, 0, 0);
#else
  // Fallback: register-staged copy (synchronous, still uses LDS for reuse).
  v8bf t = *(const v8bf*)gsrc;
  *(v8bf*)lds_dst = t;
#endif
}

__device__ inline void wait_async_le0() {
#if HAVE_WAIT_ASYNC
  __builtin_amdgcn_s_wait_asynccnt(0);
#elif HAVE_ASYNC_LDS
  asm volatile("s_wait_asynccnt 0x0" ::: "memory");
#endif
}

__device__ inline void wait_async_le1() {
#if HAVE_WAIT_ASYNC
  __builtin_amdgcn_s_wait_asynccnt(1);
#elif HAVE_ASYNC_LDS
  asm volatile("s_wait_asynccnt 0x1" ::: "memory");
#endif
}

// ---------------- weight packing (once per launch) ----------------
// Packed B layout for a KxN matrix: per 32x16 tile, half index = lane*16 + i,
// k = kt*32 + (lane>=16 ? 16:0) + i, n = nt*16 + (lane&15).

__global__ void pack_wg(const float* __restrict__ Wg, __bf16* __restrict__ out) {
  // Wg: (4, C, NQ) -> B matrix K=C(2048), N=32 (n = g*8+q). tile = nt*64 + kt.
  int idx = blockIdx.x * blockDim.x + threadIdx.x;
  if (idx >= 128 * 512) return;
  int tile = idx >> 9, r = idx & 511;
  int lane = r >> 4, i = r & 15;
  int nt = tile >> 6, kt = tile & 63;
  int k = kt * 32 + ((lane & 16) ? 16 : 0) + i;
  int n = nt * 16 + (lane & 15);
  int g = n >> 3, q = n & 7;
  out[idx] = (__bf16)Wg[(g * C + k) * NQ + q];
}

__global__ void pack_wp(const float* __restrict__ Wp, __bf16* __restrict__ out) {
  // Wp: (4, NQ, H) -> per gate: B matrix K=32 (rows 8..31 zero), N=H. tile = g*64 + nt.
  int idx = blockIdx.x * blockDim.x + threadIdx.x;
  if (idx >= 4 * 64 * 512) return;
  int tile = idx >> 9, r = idx & 511;
  int lane = r >> 4, i = r & 15;
  int g = tile >> 6, nt = tile & 63;
  int k = ((lane & 16) ? 16 : 0) + i;
  int n = nt * 16 + (lane & 15);
  float v = (k < NQ) ? Wp[(g * NQ + k) * H + n] : 0.f;
  out[idx] = (__bf16)v;
}

__global__ void pack_wtag(const float* __restrict__ Wt, __bf16* __restrict__ out) {
  // W_tag: (H, TAGS) row-major = (K,N). tile = nt*32 + kt.
  int idx = blockIdx.x * blockDim.x + threadIdx.x;
  if (idx >= 2048 * 512) return;
  int tile = idx >> 9, r = idx & 511;
  int lane = r >> 4, i = r & 15;
  int nt = tile >> 5, kt = tile & 31;
  int k = kt * 32 + ((lane & 16) ? 16 : 0) + i;
  int n = nt * 16 + (lane & 15);
  out[idx] = (__bf16)Wt[k * TAGS + n];
}

__global__ void zero_ws(unsigned int* __restrict__ p, size_t n) {
  size_t i = (size_t)blockIdx.x * blockDim.x + threadIdx.x;
  if (i < n) p[i] = 0u;
}

// ---------------- per-timestep kernels ----------------

__global__ void build_combx(const int* __restrict__ sentence,
                            const float* __restrict__ emb,
                            __bf16* __restrict__ comb, int t) {
  // comb[b][0..D-1] = bf16(emb[sentence[t][b]][d]); h-half written by step_update.
  int idx = blockIdx.x * blockDim.x + threadIdx.x;
  if (idx >= B * D) return;
  int b = idx >> 10, d = idx & 1023;
  int tok = sentence[t * B + b];
  comb[b * C + d] = (__bf16)emb[(size_t)tok * D + d];
}

__global__ void gate_qlayer(const __bf16* __restrict__ comb,
                            const __bf16* __restrict__ wgP,
                            const float* __restrict__ bg,
                            const float* __restrict__ theta,
                            __bf16* __restrict__ qpad) {
  // raw(256x32) = comb(256x2048) @ WgT(2048x32); q = cos(raw + bg + theta)
  // One wave per 16x16 output tile: grid = 16 (mt) x 2 (nt) = 32 waves.
  int lane = threadIdx.x;
  int mt = blockIdx.x >> 1, nt = blockIdx.x & 1;
  v8f acc = {};
  const __bf16* a0 = comb + (mt * 16) * C;
  const __bf16* b0 = wgP + (size_t)(nt * 64) * 512;
  for (int kt = 0; kt < 64; ++kt) {
    v16bf a = load_a_rowmajor(a0 + kt * 32, C, lane);
    v16bf b = load_b_packed(b0 + kt * 512, lane);
    acc = wmma_bf16(a, b, acc);
  }
  int gc = nt * 16 + (lane & 15);          // global gate-gemm column
  int g = gc >> 3, q = gc & 7;
  float add = bg[g * NQ + q] + theta[g * NQ + q];
  int mbase = mt * 16 + ((lane & 16) ? 8 : 0);
#pragma unroll
  for (int v = 0; v < 8; ++v) {
    float val = cosf(acc[v] + add);
    // qpad[g]: (B x 32) row-major, cols 8..31 stay zero (K-padding)
    qpad[((size_t)g * B + (mbase + v)) * GN + q] = (__bf16)val;
  }
}

__global__ void step_update(const __bf16* __restrict__ qpad,
                            const __bf16* __restrict__ wpP,
                            const float* __restrict__ bp,
                            float* __restrict__ cx,
                            __bf16* __restrict__ comb,     // writes h-half
                            __bf16* __restrict__ lstm_t) { // (B,H) bf16 record
  // For one 16x16 (b,h) tile: 4 WMMAs (one per gate, K=32 padded) + LSTM update.
  int wave = (blockIdx.x * blockDim.x + threadIdx.x) >> 5;  // 1024 waves
  int lane = threadIdx.x & 31;
  int mt = wave >> 6, nt = wave & 63;
  v8f acc[4];
#pragma unroll
  for (int g = 0; g < 4; ++g) {
    v16bf a = load_a_rowmajor(qpad + ((size_t)g * B + mt * 16) * GN, GN, lane);
    v16bf b = load_b_packed(wpP + (size_t)(g * 64 + nt) * 512, lane);
    v8f z = {};
    acc[g] = wmma_bf16(a, b, z);
  }
  int n = nt * 16 + (lane & 15);
  int mbase = mt * 16 + ((lane & 16) ? 8 : 0);
  float bpf = bp[0 * H + n], bpi = bp[1 * H + n];
  float bpg = bp[2 * H + n], bpo = bp[3 * H + n];
#pragma unroll
  for (int v = 0; v < 8; ++v) {
    int b_row = mbase + v;
    int idx = b_row * H + n;
    float fg = sigmoidf_(acc[0][v] + bpf);
    float ig = sigmoidf_(acc[1][v] + bpi);
    float gg = tanhf(acc[2][v] + bpg);
    float og = sigmoidf_(acc[3][v] + bpo);
    float c = fg * cx[idx] + ig * gg;
    cx[idx] = c;
    __bf16 hb = (__bf16)(og * tanhf(c));
    comb[b_row * C + D + n] = hb;   // feeds next step's gate gemm
    lstm_t[idx] = hb;               // record for tag projection
  }
}

// ---------------- final projection + batch-axis log-softmax ----------------

__global__ __launch_bounds__(256)
void tag_gemm(const __bf16* __restrict__ lstm,
              const __bf16* __restrict__ wtP,
              const float* __restrict__ b_tag,
              float* __restrict__ out) {
  // (T*B, H) @ (H, TAGS). Block = 8 waves -> 128x64 output tile.
  // B slab (4 n-tiles x 1KB) for the current kt is staged into LDS via
  // async global->LDS copies (double-buffered), shared by all 8 waves.
  __shared__ __align__(64) __bf16 sB[2][4 * 512];   // 2 x 4KB

  int tid  = threadIdx.x;
  int wave = tid >> 5, lane = tid & 31;
  int mb = blockIdx.x >> 4;      // 256 row-blocks of 128 rows
  int ns = blockIdx.x & 15;      // 16 col-strips of 4 n-tiles
  int mt = mb * 8 + wave;

  const __bf16* a0 = lstm + (size_t)(mt * 16) * H;

  // This thread's slice of the B slab: tile jld, 16 bytes at soff.
  int jld  = tid >> 6;           // 64 threads per 1KB tile
  int soff = (tid & 63) * 16;    // byte offset inside tile

  // issue slab kt into buffer buf
  auto issue = [&](int kt, int buf) {
    const char* g = (const char*)(wtP + (size_t)((ns * 4 + jld) * 32 + kt) * 512) + soff;
    char* l = (char*)(&sB[buf][0]) + jld * 1024 + soff;
    async_copy16(g, l);
  };

  v8f acc[4] = {};
  issue(0, 0);
  for (int kt = 0; kt < 32; ++kt) {
    int cur = kt & 1;
    if (kt + 1 < 32) {
      issue(kt + 1, cur ^ 1);
      wait_async_le1();          // my slab-kt copy is done (in-order)
    } else {
      wait_async_le0();
    }
    __syncthreads();             // everyone's slab-kt copy is done

    __builtin_prefetch(a0 + (kt + 2) * 32, 0, 1);
    v16bf a = load_a_rowmajor(a0 + kt * 32, H, lane);
#pragma unroll
    for (int j = 0; j < 4; ++j) {
      v16bf bt = *(const v16bf*)((const char*)(&sB[cur][0]) + j * 1024 + lane * 32);
      acc[j] = wmma_bf16(a, bt, acc[j]);
    }
    __syncthreads();             // reads done before buffer is overwritten
  }

  int mbase = mt * 16 + ((lane & 16) ? 8 : 0);
#pragma unroll
  for (int j = 0; j < 4; ++j) {
    int n = (ns * 4 + j) * 16 + (lane & 15);
    float bias = b_tag[n];
#pragma unroll
    for (int v = 0; v < 8; ++v)
      out[(size_t)(mbase + v) * TAGS + n] = acc[j][v] + bias;
  }
}

__global__ void log_softmax_batch(float* __restrict__ out) {
  // log_softmax over axis=1 (batch) of (T, B, TAGS); one thread per (t, tag).
  int t = blockIdx.x >> 2;
  int v = (blockIdx.x & 3) * 256 + threadIdx.x;
  float* base = out + (size_t)t * B * TAGS + v;
  float m = -INFINITY, s = 0.f;
  for (int b = 0; b < B; ++b) {
    float x = base[(size_t)b * TAGS];
    if (x > m) { s = s * __expf(m - x) + 1.f; m = x; }
    else       { s += __expf(x - m); }
  }
  float lse = m + __logf(s);
  for (int b = 0; b < B; ++b)
    base[(size_t)b * TAGS] -= lse;
}

// ---------------- launch ----------------

extern "C" void kernel_launch(void* const* d_in, const int* in_sizes, int n_in,
                              void* d_out, int out_size, void* d_ws, size_t ws_size,
                              hipStream_t stream) {
  const int*   sentence = (const int*)d_in[0];
  const float* emb    = (const float*)d_in[1];
  const float* Wg     = (const float*)d_in[2];
  const float* bg     = (const float*)d_in[3];
  const float* theta  = (const float*)d_in[4];
  const float* Wp     = (const float*)d_in[5];
  const float* bp     = (const float*)d_in[6];
  const float* W_tag  = (const float*)d_in[7];
  const float* b_tag  = (const float*)d_in[8];
  float* out = (float*)d_out;

  char* ws = (char*)d_ws;
  __bf16* lstm  = (__bf16*)ws; ws += (size_t)T * B * H * 2;   // 64 MB
  __bf16* wtP   = (__bf16*)ws; ws += (size_t)2048 * 512 * 2;  // 2 MB
  __bf16* wgP   = (__bf16*)ws; ws += (size_t)128 * 512 * 2;   // 128 KB
  __bf16* wpP   = (__bf16*)ws; ws += (size_t)4 * 64 * 512 * 2;// 256 KB
  float*  cx    = (float*)ws;  ws += (size_t)B * H * 4;       // 1 MB
  __bf16* comb  = (__bf16*)ws; ws += (size_t)B * C * 2;       // 1 MB
  __bf16* qpad  = (__bf16*)ws; ws += (size_t)4 * B * GN * 2;  // 64 KB

  // zero cx + comb + qpad (contiguous region starting at cx)
  size_t zero_words = ((size_t)B * H * 4 + (size_t)B * C * 2 + (size_t)4 * B * GN * 2) / 4;
  zero_ws<<<(unsigned)((zero_words + 255) / 256), 256, 0, stream>>>(
      (unsigned int*)cx, zero_words);

  pack_wg  <<<128 * 512 / 256, 256, 0, stream>>>(Wg, wgP);
  pack_wp  <<<4 * 64 * 512 / 256, 256, 0, stream>>>(Wp, wpP);
  pack_wtag<<<2048 * 512 / 256, 256, 0, stream>>>(W_tag, wtP);

  for (int t = 0; t < T; ++t) {
    build_combx<<<B * D / 256, 256, 0, stream>>>(sentence, emb, comb, t);
    gate_qlayer<<<32, 32, 0, stream>>>(comb, wgP, bg, theta, qpad);
    step_update<<<128, 256, 0, stream>>>(qpad, wpP, bp, cx, comb,
                                         lstm + (size_t)t * B * H);
  }

  tag_gemm<<<4096, 256, 0, stream>>>(lstm, wtP, b_tag, out);
  log_softmax_batch<<<T * 4, 256, 0, stream>>>(out);
}
